// AttnBlock_24412594110471
// MI455X (gfx1250) — matrix operand order repaired and checked by reading, not compiled, not run on hardware
//
#include <hip/hip_runtime.h>
#include <hip/hip_bf16.h>
#include <math.h>

// ---------------------------------------------------------------------------
// Types for WMMA fragments (gfx1250, wave32)
// ---------------------------------------------------------------------------
typedef __bf16 bf16raw;
typedef bf16raw v16bf __attribute__((ext_vector_type(16)));
typedef float   v8f   __attribute__((ext_vector_type(8)));
typedef unsigned int v4u __attribute__((ext_vector_type(4)));

union FragU { v16bf v; v4u u[2]; };

#define C_DIM   512
#define HW_DIM  4096
#define BATCH   2
#define GROUPS  32
#define CG      (C_DIM / GROUPS)   // 16 channels per group

// ---------------------------------------------------------------------------
// Fragment loaders (match CDNA5 ISA 16-bit A 16x32 / B 32x16 VGPR layouts)
// A (16x32, row-major source, stride lda):
//   lanes 0-15:  M=lane,    VGPR0-3 = K 0..7,  VGPR4-7 = K 16..23
//   lanes 16-31: M=lane-16, VGPR0-3 = K 8..15, VGPR4-7 = K 24..31
// ---------------------------------------------------------------------------
__device__ __forceinline__ v16bf load_frag_a(const __hip_bfloat16* A, int lda,
                                             int m0, int k0, int lane) {
  int r    = lane & 15;
  int koff = (lane >> 4) << 3;          // 0 or 8
  const __hip_bfloat16* p = A + (size_t)(m0 + r) * lda + (size_t)(k0 + koff);
  FragU f;
  f.u[0] = *reinterpret_cast<const v4u*>(p);        // K = koff .. koff+7
  f.u[1] = *reinterpret_cast<const v4u*>(p + 16);   // K = koff+16 .. koff+23
  return f.v;
}

// B (32x16) from BT (row-major [Ncols, K], stride ldb):
//   lanes 0-15:  N=lane,    VGPR0-3 = K 0..7,  VGPR4-7 = K 16..23
//   lanes 16-31: N=lane-16, VGPR0-3 = K 8..15, VGPR4-7 = K 24..31
__device__ __forceinline__ v16bf load_frag_b(const __hip_bfloat16* BT, int ldb,
                                             int n0, int k0, int lane) {
  int c    = lane & 15;
  int koff = (lane >> 4) << 3;          // 0 or 8
  const __hip_bfloat16* p = BT + (size_t)(n0 + c) * ldb + (size_t)(k0 + koff);
  FragU f;
  f.u[0] = *reinterpret_cast<const v4u*>(p);        // K = koff .. koff+7
  f.u[1] = *reinterpret_cast<const v4u*>(p + 16);   // K = koff+16 .. koff+23
  return f.v;
}

// ---------------------------------------------------------------------------
// Generic WMMA GEMM:  out = alpha * A * BT^T (+ bias) (+ residual)
//   A  : bf16 [M, K]      row-major, lda
//   BT : bf16 [N, K]      row-major, ldb   (i.e. B transposed)
//   mode 0: out bf16 [M, N] row-major (ldo)
//   mode 1: out bf16 [N, M] (transposed store; ldo = stride of the M dim)
//   mode 2: out f32  [M, N] row-major + residual f32 (same layout)
// Workgroup: 256 threads = 8 waves; tile 128(M) x 128(N); wave = 64x32.
// Requires M%128==0, N%128==0, K%32==0 (true for all launches here).
// ---------------------------------------------------------------------------
__global__ __launch_bounds__(256)
void gemm_wmma_kernel(const __hip_bfloat16* __restrict__ A, int lda,
                      const __hip_bfloat16* __restrict__ BT, int ldb,
                      const float* __restrict__ bias,
                      const float* __restrict__ residual,
                      void* __restrict__ out, int ldo,
                      int M, int N, int K, float alpha, int mode) {
  const int lane  = threadIdx.x & 31;
  const int wave  = threadIdx.x >> 5;      // 0..7
  const int waveM = (wave >> 2) * 64;      // 0 or 64
  const int waveN = (wave & 3) * 32;       // 0,32,64,96

  const int blockN = blockIdx.x * 128;
  const int blockM = blockIdx.y * 128;

  const int m_base = blockM + waveM;       // wave owns rows m_base..m_base+63
  const int n_base = blockN + waveN;       // wave owns cols n_base..n_base+31

  v8f acc[4][2];
#pragma unroll
  for (int i = 0; i < 4; ++i)
#pragma unroll
    for (int j = 0; j < 2; ++j)
      acc[i][j] = (v8f)(0.0f);

  for (int k0 = 0; k0 < K; k0 += 32) {
    v16bf afrag[4];
    v16bf bfrag[2];
#pragma unroll
    for (int i = 0; i < 4; ++i)
      afrag[i] = load_frag_a(A, lda, m_base + i * 16, k0, lane);
#pragma unroll
    for (int j = 0; j < 2; ++j)
      bfrag[j] = load_frag_b(BT, ldb, n_base + j * 16, k0, lane);

#pragma unroll
    for (int i = 0; i < 4; ++i)
#pragma unroll
      for (int j = 0; j < 2; ++j)
        acc[i][j] = __builtin_amdgcn_wmma_f32_16x16x32_bf16(
            false, afrag[i], false, bfrag[j], (short)0, acc[i][j], false, false);
  }

  // Epilogue. C/D layout: VGPR r -> row m0+r (lanes 0-15) / m0+8+r (lanes 16-31),
  // col = n0 + (lane & 15).
  const int colh = lane & 15;
  const int half = lane >> 4;

#pragma unroll
  for (int i = 0; i < 4; ++i) {
    const int m0 = m_base + i * 16;
#pragma unroll
    for (int j = 0; j < 2; ++j) {
      const int n0 = n_base + j * 16;
      const int col = n0 + colh;
      const float bv = bias ? bias[col] : 0.0f;
      v8f a = acc[i][j];
#pragma unroll
      for (int r = 0; r < 8; ++r) {
        const int row = m0 + r + half * 8;
        const float v = a[r] * alpha + bv;
        if (mode == 0) {
          ((__hip_bfloat16*)out)[(size_t)row * ldo + col] = __float2bfloat16(v);
        } else if (mode == 1) {
          ((__hip_bfloat16*)out)[(size_t)col * ldo + row] = __float2bfloat16(v);
        } else {
          const size_t idx = (size_t)row * ldo + col;
          ((float*)out)[idx] = v + residual[idx];
        }
      }
    }
  }
}

// ---------------------------------------------------------------------------
// Weight prep: w f32 [C, C] (c, o)  ->  wt bf16 [o, c]
// ---------------------------------------------------------------------------
__global__ __launch_bounds__(256)
void transpose_w_kernel(const float* __restrict__ w, __hip_bfloat16* __restrict__ wt) {
  int idx = blockIdx.x * 256 + threadIdx.x;      // idx over [o, c]
  int o = idx >> 9;            // / 512
  int c = idx & (C_DIM - 1);   // % 512
  wt[idx] = __float2bfloat16(w[(size_t)c * C_DIM + o]);
}

// ---------------------------------------------------------------------------
// GroupNorm: one block per (batch, group); reduce over HW * CG = 65536 elems.
// Writes h as bf16 [B, HW, C].
// ---------------------------------------------------------------------------
__global__ __launch_bounds__(256)
void groupnorm_kernel(const float* __restrict__ x,
                      const float* __restrict__ scale,
                      const float* __restrict__ bias,
                      __hip_bfloat16* __restrict__ h) {
  const int b = blockIdx.x / GROUPS;
  const int g = blockIdx.x % GROUPS;
  const int tid = threadIdx.x;
  const size_t base = (size_t)b * HW_DIM * C_DIM + (size_t)g * CG;

  float s = 0.0f, ss = 0.0f;
  for (int idx = tid; idx < HW_DIM * CG; idx += 256) {
    const int n = idx >> 4;          // / CG
    const int c = idx & (CG - 1);    // % CG
    const float v = x[base + (size_t)n * C_DIM + c];
    s  += v;
    ss += v * v;
  }

  __shared__ float red0[256];
  __shared__ float red1[256];
  red0[tid] = s; red1[tid] = ss;
  __syncthreads();
  for (int off = 128; off > 0; off >>= 1) {
    if (tid < off) { red0[tid] += red0[tid + off]; red1[tid] += red1[tid + off]; }
    __syncthreads();
  }
  const float inv_cnt = 1.0f / (float)(HW_DIM * CG);
  const float mean = red0[0] * inv_cnt;
  const float var  = red1[0] * inv_cnt - mean * mean;
  const float rstd = rsqrtf(var + 1e-6f);

  for (int idx = tid; idx < HW_DIM * CG; idx += 256) {
    const int n = idx >> 4;
    const int c = idx & (CG - 1);
    const int ch = g * CG + c;
    const float v = x[base + (size_t)n * C_DIM + c];
    const float hn = (v - mean) * rstd * scale[ch] + bias[ch];
    h[base + (size_t)n * C_DIM + c] = __float2bfloat16(hn);
  }
}

// ---------------------------------------------------------------------------
// Row softmax, in-place on bf16 scores. One block per row (Ncols = 4096).
// ---------------------------------------------------------------------------
__global__ __launch_bounds__(256)
void softmax_rows_kernel(__hip_bfloat16* __restrict__ S, int Ncols) {
  __hip_bfloat16* p = S + (size_t)blockIdx.x * Ncols;
  const int tid = threadIdx.x;
  __shared__ float red[256];

  float m = -3.0e38f;
  for (int i = tid; i < Ncols; i += 256)
    m = fmaxf(m, __bfloat162float(p[i]));
  red[tid] = m;
  __syncthreads();
  for (int off = 128; off > 0; off >>= 1) {
    if (tid < off) red[tid] = fmaxf(red[tid], red[tid + off]);
    __syncthreads();
  }
  m = red[0];
  __syncthreads();

  float s = 0.0f;
  for (int i = tid; i < Ncols; i += 256)
    s += __expf(__bfloat162float(p[i]) - m);
  red[tid] = s;
  __syncthreads();
  for (int off = 128; off > 0; off >>= 1) {
    if (tid < off) red[tid] += red[tid + off];
    __syncthreads();
  }
  const float inv = 1.0f / red[0];

  for (int i = tid; i < Ncols; i += 256) {
    const float e = __expf(__bfloat162float(p[i]) - m) * inv;
    p[i] = __float2bfloat16(e);
  }
}

// ---------------------------------------------------------------------------
// Host-side orchestration
// ---------------------------------------------------------------------------
extern "C" void kernel_launch(void* const* d_in, const int* in_sizes, int n_in,
                              void* d_out, int out_size, void* d_ws, size_t ws_size,
                              hipStream_t stream) {
  (void)in_sizes; (void)n_in; (void)out_size; (void)ws_size;

  const float* x      = (const float*)d_in[0];
  const float* nscale = (const float*)d_in[1];
  const float* nbias  = (const float*)d_in[2];
  const float* wq     = (const float*)d_in[3];
  const float* bq     = (const float*)d_in[4];
  const float* wk     = (const float*)d_in[5];
  const float* bk     = (const float*)d_in[6];
  const float* wv     = (const float*)d_in[7];
  const float* bv     = (const float*)d_in[8];
  const float* wproj  = (const float*)d_in[9];
  const float* bproj  = (const float*)d_in[10];
  float* outp = (float*)d_out;

  const int BN   = BATCH * HW_DIM;                 // 8192
  const size_t BNC = (size_t)BN * C_DIM;           // 4,194,304

  // Workspace carving (256-byte aligned slices)
  char* w = (char*)d_ws;
  auto alloc = [&](size_t bytes) -> char* {
    char* p = w;
    w += (bytes + 255) & ~(size_t)255;
    return p;
  };
  __hip_bfloat16* h_bf  = (__hip_bfloat16*)alloc(BNC * 2);                    // 8 MB
  __hip_bfloat16* q_bf  = (__hip_bfloat16*)alloc(BNC * 2);                    // 8 MB
  __hip_bfloat16* k_bf  = (__hip_bfloat16*)alloc(BNC * 2);                    // 8 MB
  __hip_bfloat16* vt_bf = (__hip_bfloat16*)alloc(BNC * 2);                    // 8 MB  [b][c][n]
  __hip_bfloat16* ao_bf = (__hip_bfloat16*)alloc(BNC * 2);                    // 8 MB
  __hip_bfloat16* wtq   = (__hip_bfloat16*)alloc((size_t)C_DIM * C_DIM * 2);
  __hip_bfloat16* wtk   = (__hip_bfloat16*)alloc((size_t)C_DIM * C_DIM * 2);
  __hip_bfloat16* wtv   = (__hip_bfloat16*)alloc((size_t)C_DIM * C_DIM * 2);
  __hip_bfloat16* wtp   = (__hip_bfloat16*)alloc((size_t)C_DIM * C_DIM * 2);
  __hip_bfloat16* S     = (__hip_bfloat16*)alloc((size_t)HW_DIM * HW_DIM * 2); // 32 MB, reused per batch

  const dim3 blk(256);

  // 1) Weight transposes -> bf16 W^T[o, c]
  const int wgrid = (C_DIM * C_DIM) / 256;
  transpose_w_kernel<<<wgrid, blk, 0, stream>>>(wq, wtq);
  transpose_w_kernel<<<wgrid, blk, 0, stream>>>(wk, wtk);
  transpose_w_kernel<<<wgrid, blk, 0, stream>>>(wv, wtv);
  transpose_w_kernel<<<wgrid, blk, 0, stream>>>(wproj, wtp);

  // 2) GroupNorm -> h bf16
  groupnorm_kernel<<<BATCH * GROUPS, blk, 0, stream>>>(x, nscale, nbias, h_bf);

  // 3) Q, K projections over both batches at once: [8192,512] x [512,512]
  const dim3 g_qkv(C_DIM / 128, BN / 128);   // (4, 64)
  gemm_wmma_kernel<<<g_qkv, blk, 0, stream>>>(h_bf, C_DIM, wtq, C_DIM, bq, nullptr,
                                              q_bf, C_DIM, BN, C_DIM, C_DIM, 1.0f, 0);
  gemm_wmma_kernel<<<g_qkv, blk, 0, stream>>>(h_bf, C_DIM, wtk, C_DIM, bk, nullptr,
                                              k_bf, C_DIM, BN, C_DIM, C_DIM, 1.0f, 0);

  // 4) V projection per batch with transposed store -> vt[b][c][n]
  const dim3 g_v(C_DIM / 128, HW_DIM / 128); // (4, 32)
  for (int b = 0; b < BATCH; ++b) {
    gemm_wmma_kernel<<<g_v, blk, 0, stream>>>(
        h_bf + (size_t)b * HW_DIM * C_DIM, C_DIM, wtv, C_DIM, bv, nullptr,
        vt_bf + (size_t)b * C_DIM * HW_DIM, HW_DIM, HW_DIM, C_DIM, C_DIM, 1.0f, 1);
  }

  // 5) Attention per batch (S buffer reused; kernels serialize on the stream)
  const float scale = 1.0f / sqrtf((float)C_DIM); // C^-0.5
  const dim3 g_s(HW_DIM / 128, HW_DIM / 128);     // (32, 32)
  const dim3 g_pv(C_DIM / 128, HW_DIM / 128);     // (4, 32)
  for (int b = 0; b < BATCH; ++b) {
    const size_t off = (size_t)b * HW_DIM * C_DIM;
    // S = scale * Q K^T  (A=q [4096,512], BT=k [4096,512])
    gemm_wmma_kernel<<<g_s, blk, 0, stream>>>(q_bf + off, C_DIM, k_bf + off, C_DIM,
                                              nullptr, nullptr, S, HW_DIM,
                                              HW_DIM, HW_DIM, C_DIM, scale, 0);
    // softmax rows, in place
    softmax_rows_kernel<<<HW_DIM, blk, 0, stream>>>(S, HW_DIM);
    // O = P V  (A=P [4096,4096], BT=vt[b] [512,4096])
    gemm_wmma_kernel<<<g_pv, blk, 0, stream>>>(S, HW_DIM,
                                               vt_bf + (size_t)b * C_DIM * HW_DIM, HW_DIM,
                                               nullptr, nullptr, ao_bf + off, C_DIM,
                                               HW_DIM, C_DIM, HW_DIM, 1.0f, 0);
  }

  // 6) Output projection + bias + residual, f32 out
  gemm_wmma_kernel<<<g_qkv, blk, 0, stream>>>(ao_bf, C_DIM, wtp, C_DIM, bproj, x,
                                              outp, C_DIM, BN, C_DIM, C_DIM, 1.0f, 2);
}
